// SagPoolGAT_66022237274494
// MI455X (gfx1250) — matrix operand order repaired and checked by reading, not compiled, not run on hardware
//
#include <hip/hip_runtime.h>
#include <hip/hip_fp16.h>
#include <math.h>

// ---------------- problem constants (match reference) ----------------
#define NNODES 65536
#define BGR    16
#define NPG0   4096
#define NEDGE  524288
#define FIN    128
#define HIDF   64
#define NHEADS 4
#define HFD    256
#define NCLS   10
#define SLOPE  0.2f

#define CDIV(a, b) (int)(((long)(a) + (b) - 1) / (b))

typedef __attribute__((ext_vector_type(16))) _Float16 v16h;
typedef __attribute__((ext_vector_type(8)))  float    v8f;

union Frag {
  v16h  v;
  uint4 q[2];
};

__device__ __forceinline__ float leakyr(float x) { return x > 0.f ? x : SLOPE * x; }

__device__ __forceinline__ void atomicMaxF32(float* addr, float v) {
  unsigned int* ua  = (unsigned int*)addr;
  unsigned int  old = *(volatile unsigned int*)ua;
  while (__uint_as_float(old) < v) {
    unsigned int assumed = old;
    old = atomicCAS(ua, assumed, __float_as_uint(v));
    if (old == assumed) break;
  }
}

// ---------------- utility kernels ----------------
__global__ void fill_f32(float* p, float v, long n) {
  long t = (long)blockIdx.x * blockDim.x + threadIdx.x;
  if (t < n) p[t] = v;
}
__global__ void fill_i32(int* p, int v, long n) {
  long t = (long)blockIdx.x * blockDim.x + threadIdx.x;
  if (t < n) p[t] = v;
}

// ---------------- WMMA fragment packing (f32 -> f16, fragment-native layout) ---------
// A[M,K] -> tiles of 16 rows x 32 k. Tile (tileR, tileK) stored as 32 lanes x 16 halves
// contiguous, per CDNA5 ISA "16-bit A-Matrix 16x32" layout.
__global__ void pack_a_f16(const float* __restrict__ A, unsigned short* __restrict__ P,
                           int M, int K) {
  long t = (long)blockIdx.x * blockDim.x + threadIdx.x;
  if (t >= (long)M * K) return;
  int idx  = (int)(t & 511);          // element within 512-half tile
  long tile = t >> 9;
  int KT   = K >> 5;
  int tileK = (int)(tile % KT);
  int tileR = (int)(tile / KT);
  int L = idx >> 4, j = idx & 15;
  int m = L & 15, hh = L >> 4;
  int p = j >> 1, lo = j & 1;
  int kb = ((p & 4) << 2) + hh * 8 + ((p & 3) << 1) + lo;
  float v = A[(size_t)(tileR * 16 + m) * K + tileK * 32 + kb];
  P[t] = __half_as_ushort(__float2half(v));
}

// B[K,Nc] -> tiles of 32 k x 16 n, tile order (tileN, tileK) so the GEMM k-loop for a
// fixed column tile streams contiguously. Lane = N column, halves j -> K per ISA layout.
__global__ void pack_b_f16(const float* __restrict__ B, unsigned short* __restrict__ P,
                           int K, int Nc) {
  long t = (long)blockIdx.x * blockDim.x + threadIdx.x;
  if (t >= (long)K * Nc) return;
  int idx  = (int)(t & 511);
  long tile = t >> 9;
  int KT   = K >> 5;
  int tileK = (int)(tile % KT);
  int tileN = (int)(tile / KT);
  int L = idx >> 4, j = idx & 15;
  int n = L & 15, hh = L >> 4;
  int p = j >> 1, lo = j & 1;
  int kk = ((p & 4) << 2) + hh * 8 + ((p & 3) << 1) + lo;
  float v = B[(size_t)(tileK * 32 + kk) * Nc + tileN * 16 + n];
  P[t] = __half_as_ushort(__float2half(v));
}

// ---------------- WMMA GEMM on packed operands ----------------
// C[M,256] = A[M,K] * B[K,256]; block = 256 threads = 8 waves (2 row-tiles x 4 col-groups);
// each wave owns 16 rows x 64 cols (4 accumulators). Grid.x = M/32.
// Per lane per fragment: 2 x global_load_b128 (fully coalesced).
__global__ void wmma_gemm_f16(const unsigned short* __restrict__ Ap,
                              const unsigned short* __restrict__ Bp,
                              float* __restrict__ C, int M, int K, int Nc) {
  const int lane  = threadIdx.x & 31;
  const int w     = threadIdx.x >> 5;
  const int wrow  = w >> 2, wcol = w & 3;
  const int tileR = blockIdx.x * 2 + wrow;
  const int KT    = K >> 5;

  v8f acc0 = {}, acc1 = {}, acc2 = {}, acc3 = {};

  // uint4 units: one tile = 512 halves = 64 uint4; lane offset = lane*2 uint4.
  const uint4* abase = (const uint4*)Ap + (size_t)tileR * KT * 64 + lane * 2;
  const uint4* bbase = (const uint4*)Bp + lane * 2;

  for (int kt = 0; kt < KT; ++kt) {
    Frag af;
    af.q[0] = abase[(size_t)kt * 64];
    af.q[1] = abase[(size_t)kt * 64 + 1];
    if (kt + 1 < KT) __builtin_prefetch(abase + (size_t)(kt + 1) * 64, 0, 1);
#pragma unroll
    for (int t = 0; t < 4; ++t) {
      const int tileN = wcol * 4 + t;
      const uint4* bp = bbase + ((size_t)tileN * KT + kt) * 64;
      Frag bf;
      bf.q[0] = bp[0];
      bf.q[1] = bp[1];
      v8f* ap = (t == 0) ? &acc0 : (t == 1) ? &acc1 : (t == 2) ? &acc2 : &acc3;
      *ap = __builtin_amdgcn_wmma_f32_16x16x32_f16(false, af.v, false, bf.v,
                                                   (short)0, *ap, false, false);
    }
  }

  // C/D f32 16x16 layout: lane n = col, VGPR r -> row (lane>=16 ? 8+r : r)
  const int n  = lane & 15;
  const int mo = (lane >> 4) * 8;
  const int mrow = tileR * 16;
#pragma unroll
  for (int t = 0; t < 4; ++t) {
    v8f a = (t == 0) ? acc0 : (t == 1) ? acc1 : (t == 2) ? acc2 : acc3;
    const int col = wcol * 64 + t * 16 + n;
#pragma unroll
    for (int r = 0; r < 8; ++r)
      C[(size_t)(mrow + mo + r) * Nc + col] = a[r];
  }
}

// ---------------- GAT kernels ----------------
__global__ void att_logits(const float* __restrict__ h, const float* __restrict__ asrc,
                           const float* __restrict__ adst, float* __restrict__ als,
                           float* __restrict__ ald, int nNodes) {
  int t = blockIdx.x * blockDim.x + threadIdx.x;
  if (t >= nNodes * NHEADS) return;
  int n0 = t >> 2, hd = t & 3;
  const float* hr = h + (size_t)n0 * HFD + hd * HIDF;
  const float* as = asrc + hd * HIDF;
  const float* ad = adst + hd * HIDF;
  float ss = 0.f, dd = 0.f;
  for (int f = 0; f < HIDF; ++f) { float v = hr[f]; ss += v * as[f]; dd += v * ad[f]; }
  als[t] = ss; ald[t] = dd;
}

__global__ void self_max(const float* als, const float* ald, float* emax, int nNodes) {
  int t = blockIdx.x * blockDim.x + threadIdx.x;
  if (t >= nNodes * NHEADS) return;
  emax[t] = leakyr(als[t] + ald[t]);
}

__global__ void edge_max(const int* __restrict__ src, const int* __restrict__ dst,
                         const int* __restrict__ msk, const float* __restrict__ als,
                         const float* __restrict__ ald, float* __restrict__ emax, int nE) {
  int t = blockIdx.x * blockDim.x + threadIdx.x;
  if (t >= nE * NHEADS) return;
  int e = t >> 2, hd = t & 3;
  if (msk && !msk[e]) return;
  int s = src[e], d = dst[e];
  float ev = leakyr(als[s * NHEADS + hd] + ald[d * NHEADS + hd]);
  atomicMaxF32(&emax[d * NHEADS + hd], ev);
}

__global__ void self_sum(const float* als, const float* ald, const float* emax,
                         float* esum, int nNodes) {
  int t = blockIdx.x * blockDim.x + threadIdx.x;
  if (t >= nNodes * NHEADS) return;
  esum[t] = expf(leakyr(als[t] + ald[t]) - emax[t]);
}

__global__ void edge_sum(const int* __restrict__ src, const int* __restrict__ dst,
                         const int* __restrict__ msk, const float* __restrict__ als,
                         const float* __restrict__ ald, const float* __restrict__ emax,
                         float* __restrict__ esum, int nE) {
  int t = blockIdx.x * blockDim.x + threadIdx.x;
  if (t >= nE * NHEADS) return;
  int e = t >> 2, hd = t & 3;
  if (msk && !msk[e]) return;
  int s = src[e], d = dst[e];
  float ev = leakyr(als[s * NHEADS + hd] + ald[d * NHEADS + hd]);
  atomicAdd(&esum[d * NHEADS + hd], expf(ev - emax[d * NHEADS + hd]));
}

__global__ void out_init(const float* __restrict__ h, const float* __restrict__ als,
                         const float* __restrict__ ald, const float* __restrict__ emax,
                         const float* __restrict__ esum, const float* __restrict__ bias,
                         float* __restrict__ out, int nNodes) {
  long t = (long)blockIdx.x * blockDim.x + threadIdx.x;
  if (t >= (long)nNodes * HFD) return;
  int n0 = (int)(t >> 8), c = (int)(t & 255), hd = c >> 6;
  int nh = n0 * NHEADS + hd;
  float a = expf(leakyr(als[nh] + ald[nh]) - emax[nh]) / (esum[nh] + 1e-16f);
  out[t] = bias[c] + a * h[t];
}

// wave-per-edge: out[dst,c] += alpha[h(c)] * hfeat[src,c]
__global__ void edge_agg(const int* __restrict__ src, const int* __restrict__ dst,
                         const int* __restrict__ msk, const float* __restrict__ als,
                         const float* __restrict__ ald, const float* __restrict__ emax,
                         const float* __restrict__ esum, const float* __restrict__ h,
                         float* __restrict__ out, int nE) {
  int e = blockIdx.x * (blockDim.x >> 5) + (threadIdx.x >> 5);
  if (e >= nE) return;
  if (msk && !msk[e]) return;
  int lane = threadIdx.x & 31;
  int s = src[e], d = dst[e];
#pragma unroll
  for (int i = 0; i < 8; ++i) {
    int c  = i * 32 + lane;
    int hd = c >> 6;
    int nh = d * NHEADS + hd;
    float ev = leakyr(als[s * NHEADS + hd] + ald[nh]);
    float a  = expf(ev - emax[nh]) / (esum[nh] + 1e-16f);
    atomicAdd(&out[(size_t)d * HFD + c], a * h[(size_t)s * HFD + c]);
  }
}

// ---------------- SAGPool kernels ----------------
__global__ void pool_agg(const int* __restrict__ src, const int* __restrict__ dst,
                         const int* __restrict__ msk, const float* __restrict__ x,
                         float* __restrict__ agg, int nE) {
  int e = blockIdx.x * (blockDim.x >> 5) + (threadIdx.x >> 5);
  if (e >= nE) return;
  if (msk && !msk[e]) return;
  int lane = threadIdx.x & 31;
  int s = src[e], d = dst[e];
#pragma unroll
  for (int i = 0; i < 8; ++i) {
    int c = i * 32 + lane;
    atomicAdd(&agg[(size_t)d * HFD + c], x[(size_t)s * HFD + c]);
  }
}

__global__ void score_k(const float* __restrict__ agg, const float* __restrict__ x,
                        const float* __restrict__ relw, const float* __restrict__ relb,
                        const float* __restrict__ rootw, float* __restrict__ score, int nNodes) {
  int n0 = blockIdx.x * blockDim.x + threadIdx.x;
  if (n0 >= nNodes) return;
  float sc = relb[0];
  const float* ar = agg + (size_t)n0 * HFD;
  const float* xr = x + (size_t)n0 * HFD;
  for (int c = 0; c < HFD; ++c) sc += ar[c] * relw[c] + xr[c] * rootw[c];
  score[n0] = sc;
}

// one workgroup per graph; bitonic sort (score desc) of npg entries in LDS; keep top k
__global__ void topk_kernel(const float* __restrict__ score, int* __restrict__ perm,
                            int* __restrict__ kept, int* __restrict__ nidx, int npg, int k) {
  __shared__ float ss[4096];
  __shared__ int   si[4096];
  const int g = blockIdx.x;
  for (int i = threadIdx.x; i < npg; i += blockDim.x) { ss[i] = score[g * npg + i]; si[i] = i; }
  __syncthreads();
  for (unsigned ksz = 2; ksz <= (unsigned)npg; ksz <<= 1) {
    for (unsigned j = ksz >> 1; j > 0; j >>= 1) {
      for (int i = threadIdx.x; i < npg; i += blockDim.x) {
        int ixj = i ^ (int)j;
        if (ixj > i) {
          bool up = ((i & ksz) == 0);
          float a = ss[i], b = ss[ixj];
          bool sw = up ? (a < b) : (a > b);   // descending
          if (sw) { ss[i] = b; ss[ixj] = a; int tt = si[i]; si[i] = si[ixj]; si[ixj] = tt; }
        }
      }
      __syncthreads();
    }
  }
  for (int i = threadIdx.x; i < k; i += blockDim.x) {
    int oldg = g * npg + si[i];
    int nn   = g * k + i;
    perm[nn]   = oldg;
    kept[oldg] = 1;
    nidx[oldg] = nn;
  }
}

// x_new = relu(x[perm] * tanh(score[perm]))   (relu of the following layer is fused)
__global__ void build_pooled(const float* __restrict__ x, const int* __restrict__ perm,
                             const float* __restrict__ score, float* __restrict__ xout, int nNew) {
  long t = (long)blockIdx.x * blockDim.x + threadIdx.x;
  if (t >= (long)nNew * HFD) return;
  int nn = (int)(t >> 8), c = (int)(t & 255);
  int oldg = perm[nn];
  float v = x[(size_t)oldg * HFD + c] * tanhf(score[oldg]);
  xout[t] = fmaxf(v, 0.f);
}

__global__ void remap_edges(const int* __restrict__ src, const int* __restrict__ dst,
                            const int* __restrict__ mskOld, const int* __restrict__ kept,
                            const int* __restrict__ nidx, int* __restrict__ sOut,
                            int* __restrict__ dOut, int* __restrict__ mOut, int nE) {
  int e = blockIdx.x * blockDim.x + threadIdx.x;
  if (e >= nE) return;
  int s = src[e], d = dst[e];
  int mk = (mskOld ? mskOld[e] : 1) & kept[s] & kept[d];
  sOut[e] = nidx[s]; dOut[e] = nidx[d]; mOut[e] = mk;
}

// ---------------- readout ----------------
__global__ void mean_pool(const float* __restrict__ x, float* __restrict__ g, int rowsPerGraph) {
  int t = blockIdx.x * blockDim.x + threadIdx.x;
  if (t >= BGR * HFD) return;
  int b = t >> 8, c = t & 255;
  float s = 0.f;
  for (int r = 0; r < rowsPerGraph; ++r) s += x[((size_t)b * rowsPerGraph + r) * HFD + c];
  g[t] = s / (float)rowsPerGraph;
}

__global__ void fc_kernel(const float* __restrict__ in, const float* __restrict__ w,
                          const float* __restrict__ b, float* __restrict__ out,
                          int M, int K, int Nc, int doRelu) {
  int t = blockIdx.x * blockDim.x + threadIdx.x;
  if (t >= M * Nc) return;
  int row = t / Nc, cn = t % Nc;
  float s = b[cn];
  for (int c = 0; c < K; ++c) s += in[row * K + c] * w[c * Nc + cn];
  out[t] = doRelu ? fmaxf(s, 0.f) : s;
}

__global__ void logsoftmax_k(const float* __restrict__ lg, float* __restrict__ out) {
  int b = threadIdx.x;
  if (b >= BGR) return;
  float mx = -1e30f;
  for (int j = 0; j < NCLS; ++j) mx = fmaxf(mx, lg[b * NCLS + j]);
  float se = 0.f;
  for (int j = 0; j < NCLS; ++j) se += expf(lg[b * NCLS + j] - mx);
  float ls = logf(se);
  for (int j = 0; j < NCLS; ++j) out[b * NCLS + j] = lg[b * NCLS + j] - mx - ls;
}

// ---------------- host-side layer drivers ----------------
static void run_gat(const unsigned short* xp, const unsigned short* Wp, const float* asrc,
                    const float* adst, const float* bias, const int* esrc, const int* edst,
                    const int* emask, int nNodes, int K, float* hbuf, float* obuf,
                    float* als, float* ald, float* emax, float* esum, hipStream_t stream) {
  wmma_gemm_f16<<<nNodes / 32, 256, 0, stream>>>(xp, Wp, hbuf, nNodes, K, HFD);
  att_logits<<<CDIV(nNodes * NHEADS, 256), 256, 0, stream>>>(hbuf, asrc, adst, als, ald, nNodes);
  self_max<<<CDIV(nNodes * NHEADS, 256), 256, 0, stream>>>(als, ald, emax, nNodes);
  edge_max<<<CDIV(NEDGE * NHEADS, 256), 256, 0, stream>>>(esrc, edst, emask, als, ald, emax, NEDGE);
  self_sum<<<CDIV(nNodes * NHEADS, 256), 256, 0, stream>>>(als, ald, emax, esum, nNodes);
  edge_sum<<<CDIV(NEDGE * NHEADS, 256), 256, 0, stream>>>(esrc, edst, emask, als, ald, emax, esum, NEDGE);
  out_init<<<CDIV((long)nNodes * HFD, 256), 256, 0, stream>>>(hbuf, als, ald, emax, esum, bias, obuf, nNodes);
  edge_agg<<<CDIV(NEDGE, 8), 256, 0, stream>>>(esrc, edst, emask, als, ald, emax, esum, hbuf, obuf, NEDGE);
}

static void run_pool(const float* xin, const int* esrc, const int* edst, const int* emask,
                     const float* relw, const float* relb, const float* rootw, int nNodes,
                     int npg, float* aggbuf, float* score, int* perm, int* kept, int* nidx,
                     float* xout, hipStream_t stream) {
  fill_f32<<<CDIV((long)nNodes * HFD, 256), 256, 0, stream>>>(aggbuf, 0.f, (long)nNodes * HFD);
  pool_agg<<<CDIV(NEDGE, 8), 256, 0, stream>>>(esrc, edst, emask, xin, aggbuf, NEDGE);
  score_k<<<CDIV(nNodes, 256), 256, 0, stream>>>(aggbuf, xin, relw, relb, rootw, score, nNodes);
  fill_i32<<<CDIV(nNodes, 256), 256, 0, stream>>>(kept, 0, nNodes);
  fill_i32<<<CDIV(nNodes, 256), 256, 0, stream>>>(nidx, 0, nNodes);
  topk_kernel<<<BGR, 1024, 0, stream>>>(score, perm, kept, nidx, npg, npg / 2);
  int nNew = nNodes / 2;
  build_pooled<<<CDIV((long)nNew * HFD, 256), 256, 0, stream>>>(xin, perm, score, xout, nNew);
}

extern "C" void kernel_launch(void* const* d_in, const int* in_sizes, int n_in,
                              void* d_out, int out_size, void* d_ws, size_t ws_size,
                              hipStream_t stream) {
  const float* x      = (const float*)d_in[0];
  const int*   src    = (const int*)d_in[1];
  const int*   dst    = (const int*)d_in[2];
  const float* W1     = (const float*)d_in[3];
  const float* a_src1 = (const float*)d_in[4];
  const float* a_dst1 = (const float*)d_in[5];
  const float* b1     = (const float*)d_in[6];
  const float* rel1_w = (const float*)d_in[7];
  const float* rel1_b = (const float*)d_in[8];
  const float* root1w = (const float*)d_in[9];
  const float* W2     = (const float*)d_in[10];
  const float* a_src2 = (const float*)d_in[11];
  const float* a_dst2 = (const float*)d_in[12];
  const float* b2     = (const float*)d_in[13];
  const float* rel2_w = (const float*)d_in[14];
  const float* rel2_b = (const float*)d_in[15];
  const float* root2w = (const float*)d_in[16];
  const float* fc1_w  = (const float*)d_in[17];
  const float* fc1_b  = (const float*)d_in[18];
  const float* fc2_w  = (const float*)d_in[19];
  const float* fc2_b  = (const float*)d_in[20];
  float* outp = (float*)d_out;

  // workspace bump allocator (deterministic layout every call)
  char*  ws  = (char*)d_ws;
  size_t off = 0;
  auto alloc = [&](size_t bytes) -> char* {
    char* p = ws + off;
    off += (bytes + 255) & ~(size_t)255;
    return p;
  };
  unsigned short* xp  = (unsigned short*)alloc((size_t)NNODES * FIN * 2);   // packed A (layer1)
  unsigned short* W1p = (unsigned short*)alloc((size_t)FIN * HFD * 2);      // packed B
  unsigned short* W2p = (unsigned short*)alloc((size_t)HFD * HFD * 2);
  unsigned short* x2p = (unsigned short*)alloc((size_t)(NNODES / 2) * HFD * 2);
  float* big0  = (float*)alloc((size_t)NNODES * HFD * 4);       // h1 / agg1 / h2 / agg2
  float* big1  = (float*)alloc((size_t)NNODES * HFD * 4);       // out1 / out2
  float* x2    = (float*)alloc((size_t)(NNODES / 2) * HFD * 4);
  float* x3    = (float*)alloc((size_t)(NNODES / 4) * HFD * 4);
  float* als   = (float*)alloc((size_t)NNODES * NHEADS * 4);
  float* ald   = (float*)alloc((size_t)NNODES * NHEADS * 4);
  float* emax  = (float*)alloc((size_t)NNODES * NHEADS * 4);
  float* esum  = (float*)alloc((size_t)NNODES * NHEADS * 4);
  float* score = (float*)alloc((size_t)NNODES * 4);
  int*   perm  = (int*)alloc((size_t)(NNODES / 2) * 4);
  int*   kept  = (int*)alloc((size_t)NNODES * 4);
  int*   nidx  = (int*)alloc((size_t)NNODES * 4);
  int*   s1e   = (int*)alloc((size_t)NEDGE * 4);
  int*   d1e   = (int*)alloc((size_t)NEDGE * 4);
  int*   m1    = (int*)alloc((size_t)NEDGE * 4);
  float* gbuf  = (float*)alloc((size_t)BGR * HFD * 4);
  float* g2buf = (float*)alloc((size_t)BGR * HIDF * 4);
  float* lgbuf = (float*)alloc((size_t)BGR * NCLS * 4);

  // ---- pack GEMM operands (f32 -> f16, WMMA fragment layout) ----
  pack_a_f16<<<CDIV((long)NNODES * FIN, 256), 256, 0, stream>>>(x, xp, NNODES, FIN);
  pack_b_f16<<<CDIV(FIN * HFD, 256), 256, 0, stream>>>(W1, W1p, FIN, HFD);
  pack_b_f16<<<CDIV(HFD * HFD, 256), 256, 0, stream>>>(W2, W2p, HFD, HFD);

  // ---- layer 1: GAT + SAGPool ----
  run_gat(xp, W1p, a_src1, a_dst1, b1, src, dst, nullptr, NNODES, FIN,
          big0, big1, als, ald, emax, esum, stream);
  run_pool(big1, src, dst, nullptr, rel1_w, rel1_b, root1w, NNODES, NPG0,
           big0, score, perm, kept, nidx, x2, stream);
  remap_edges<<<CDIV(NEDGE, 256), 256, 0, stream>>>(src, dst, nullptr, kept, nidx,
                                                    s1e, d1e, m1, NEDGE);

  // ---- layer 2: GAT + SAGPool ----
  int N2 = NNODES / 2;
  pack_a_f16<<<CDIV((long)N2 * HFD, 256), 256, 0, stream>>>(x2, x2p, N2, HFD);
  run_gat(x2p, W2p, a_src2, a_dst2, b2, s1e, d1e, m1, N2, HFD,
          big0, big1, als, ald, emax, esum, stream);
  run_pool(big1, s1e, d1e, m1, rel2_w, rel2_b, root2w, N2, NPG0 / 2,
           big0, score, perm, kept, nidx, x3, stream);

  // ---- readout: mean pool -> fc1(relu) -> fc2 -> log_softmax ----
  mean_pool<<<CDIV(BGR * HFD, 256), 256, 0, stream>>>(x3, gbuf, NPG0 / 4);
  fc_kernel<<<CDIV(BGR * HIDF, 256), 256, 0, stream>>>(gbuf, fc1_w, fc1_b, g2buf,
                                                       BGR, HFD, HIDF, 1);
  fc_kernel<<<1, 256, 0, stream>>>(g2buf, fc2_w, fc2_b, lgbuf, BGR, HIDF, NCLS, 0);
  logsoftmax_k<<<1, 32, 0, stream>>>(lgbuf, outp);
}
